// AttentivePooling_56710748176720
// MI455X (gfx1250) — compile-verified, matrix operand order checked by reading, not run on hardware
//
#include <hip/hip_runtime.h>
#include <hip/hip_bf16.h>
#include <math.h>

typedef __attribute__((ext_vector_type(16))) __bf16          v16bf;
typedef __attribute__((ext_vector_type(16))) unsigned short  v16us;
typedef __attribute__((ext_vector_type(8)))  unsigned short  v8us;
typedef __attribute__((ext_vector_type(8)))  float           v8f;

#define HIDDEN 1024
#define PROJ   175
#define PPAD   192   // 6 x K=32 WMMA chunks; cols 175..191 zero-padded
#define NB     16
#define L      2048

// ---------- helpers ----------
__device__ __forceinline__ unsigned short f2bf(float f) {
  unsigned int u = __float_as_uint(f);
  u += 0x7FFFu + ((u >> 16) & 1u);          // round-to-nearest-even
  return (unsigned short)(u >> 16);
}

// float atomic max via sign-aware int/uint punning (works with -inf init)
__device__ __forceinline__ void atomicMaxF(float* a, float v) {
  if (v >= 0.f) atomicMax((int*)a, __float_as_int(v));
  else          atomicMin((unsigned int*)a, __float_as_uint(v));
}

// A/B fragment loader for v_wmma_f32_16x16x32_bf16, per ISA 7.12.2 wave32 layout:
// lanes 0-15 : elems 0..7 = K[kb+0..7],   elems 8..15 = K[kb+16..23]
// lanes 16-31: elems 0..7 = K[kb+8..15],  elems 8..15 = K[kb+24..31]
__device__ __forceinline__ v16bf load_frag_bf16(const unsigned short* rowptr,
                                                int kb, int koff) {
  const uint4* p0 = (const uint4*)(rowptr + kb + koff);
  const uint4* p1 = (const uint4*)(rowptr + kb + koff + 16);
  v8us lo = __builtin_bit_cast(v8us, *p0);
  v8us hi = __builtin_bit_cast(v8us, *p1);
  v16us t = __builtin_shufflevector(lo, hi, 0,1,2,3,4,5,6,7,8,9,10,11,12,13,14,15);
  return __builtin_bit_cast(v16bf, t);
}

// Same fragment but converting from f32 source on the fly
__device__ __forceinline__ v16bf load_frag_f32(const float* rowptr, int kb, int koff) {
  v16us t;
#pragma unroll
  for (int i = 0; i < 8; ++i) t[i]     = f2bf(rowptr[kb + koff + i]);
#pragma unroll
  for (int i = 0; i < 8; ++i) t[8 + i] = f2bf(rowptr[kb + koff + 16 + i]);
  return __builtin_bit_cast(v16bf, t);
}

__device__ __forceinline__ v8f wmma_bf16(v16bf a, v16bf b, v8f c) {
  return __builtin_amdgcn_wmma_f32_16x16x32_bf16(false, a, false, b,
                                                 (short)0, c, false, false);
}

// ---------- staging kernels ----------
// param [1024,175] f32  ->  paramT [192,1024] bf16 (rows >=175 zero)
__global__ void k_paramT(const float* __restrict__ param, unsigned short* __restrict__ pT) {
  int idx = blockIdx.x * 256 + threadIdx.x;      // 192*1024 threads
  int p = idx >> 10, h = idx & 1023;
  pT[idx] = (p < PROJ) ? f2bf(param[h * PROJ + p]) : (unsigned short)0;
}

// second [B,2048,175] f32 -> secb [B,2048,192] bf16 (cols >=175 zero)
__global__ void k_sec(const float* __restrict__ second, unsigned short* __restrict__ secb) {
  size_t idx = (size_t)blockIdx.x * 256 + threadIdx.x;  // B*L*192 threads
  int c = (int)(idx % PPAD);
  size_t r = idx / PPAD;
  secb[idx] = (c < PROJ) ? f2bf(second[r * PROJ + c]) : (unsigned short)0;
}

__global__ void k_init_neginf(float* __restrict__ p, int n) {
  int i = blockIdx.x * 256 + threadIdx.x;
  if (i < n) p[i] = -INFINITY;
}

// ---------- proj = first @ param  (bf16 WMMA, f32 acc), 4 n-tiles/wave ------
__global__ __launch_bounds__(256)
void k_proj(const float* __restrict__ first,
            const unsigned short* __restrict__ pT,
            unsigned short* __restrict__ projb) {
  int w    = blockIdx.x * 8 + (threadIdx.x >> 5);
  int lane = threadIdx.x & 31;
  int ng = w % 3;                       // n-tile group: nt = ng*4 + u
  int it = (w / 3) % (L / 16);
  int b  = w / (3 * (L / 16));
  int row  = lane & 15;
  int koff = (lane >> 4) * 8;

  const float* arow = first + ((size_t)b * L + it * 16 + row) * HIDDEN;
  const unsigned short* brow[4];
#pragma unroll
  for (int u = 0; u < 4; ++u)
    brow[u] = pT + (size_t)((ng * 4 + u) * 16 + row) * HIDDEN;

  v8f acc[4] = {};
  for (int kb = 0; kb < HIDDEN; kb += 32) {
    v16bf a = load_frag_f32(arow, kb, koff);   // convert once, reuse 4x
#pragma unroll
    for (int u = 0; u < 4; ++u) {
      v16bf bb = load_frag_bf16(brow[u], kb, koff);
      acc[u] = wmma_bf16(a, bb, acc[u]);       // 4 independent chains
    }
  }
  // D layout: lanes 0-15 rows 0..7, lanes 16-31 rows 8..15; col = lane&15
  int half = lane >> 4;
#pragma unroll
  for (int u = 0; u < 4; ++u) {
    int col = (ng * 4 + u) * 16 + (lane & 15);
    size_t base = ((size_t)b * L + it * 16 + half * 8) * PPAD + col;
#pragma unroll
    for (int r = 0; r < 8; ++r)
      projb[base + (size_t)r * PPAD] = f2bf(acc[u][r]);
  }
}

// ---------- score strip: 16x64 tile per wave, maxes via shuffles + atomics ---
__global__ __launch_bounds__(256)
void k_score(const unsigned short* __restrict__ projb,
             const unsigned short* __restrict__ secb,
             float* __restrict__ attn_i, float* __restrict__ attn_j) {
  int w    = blockIdx.x * 8 + (threadIdx.x >> 5);
  int lane = threadIdx.x & 31;
  int tjg = w % (L / 64);                    // group of 4 tj tiles
  int ti  = (w / (L / 64)) % (L / 16);
  int b   = w / ((L / 64) * (L / 16));
  int row  = lane & 15;
  int koff = (lane >> 4) * 8;

  const unsigned short* arow = projb + ((size_t)b * L + ti * 16 + row) * PPAD;
  const unsigned short* brow[4];
#pragma unroll
  for (int u = 0; u < 4; ++u)
    brow[u] = secb + ((size_t)b * L + (tjg * 4 + u) * 16 + row) * PPAD;

  v8f acc[4] = {};
#pragma unroll
  for (int k = 0; k < PPAD / 32; ++k) {
    v16bf a = load_frag_bf16(arow, k * 32, koff);   // load once, reuse 4x
#pragma unroll
    for (int u = 0; u < 4; ++u) {
      v16bf bb = load_frag_bf16(brow[u], k * 32, koff);
      acc[u] = wmma_bf16(a, bb, acc[u]);            // 4 independent chains
    }
  }

  int half = lane >> 4;
  int l15  = lane & 15;

  // Col maxes per tj tile: element r of lane = S[half*8+r][l15]
#pragma unroll
  for (int u = 0; u < 4; ++u) {
    float cm = acc[u][0];
#pragma unroll
    for (int r = 1; r < 8; ++r) cm = fmaxf(cm, acc[u][r]);
    cm = fmaxf(cm, __shfl_xor(cm, 16, 32));         // combine row halves
    if (half == 0)
      atomicMaxF(&attn_j[(size_t)b * L + (tjg * 4 + u) * 16 + l15], cm);
  }

  // Row maxes: combine the 4 tiles locally, then reduce the 16 columns
  float rm[8];
#pragma unroll
  for (int r = 0; r < 8; ++r) {
    float v = acc[0][r];
#pragma unroll
    for (int u = 1; u < 4; ++u) v = fmaxf(v, acc[u][r]);
    rm[r] = v;
  }
#pragma unroll
  for (int m = 1; m < 16; m <<= 1)
#pragma unroll
    for (int r = 0; r < 8; ++r)
      rm[r] = fmaxf(rm[r], __shfl_xor(rm[r], m, 32));
  if (l15 < 8)
    atomicMaxF(&attn_i[(size_t)b * L + ti * 16 + half * 8 + l15], rm[l15]);
}

// ---------- softmax over 2048 per batch (tanh(max) -> softmax) ----------
__global__ __launch_bounds__(256)
void k_softmax(const float* __restrict__ attn, float* __restrict__ wout) {
  __shared__ float red[8];
  int b = blockIdx.x, t = threadIdx.x;
  const float* a = attn + (size_t)b * L;
  float vals[8];
  float mx = -INFINITY;
#pragma unroll
  for (int k = 0; k < 8; ++k) { vals[k] = tanhf(a[t + k * 256]); mx = fmaxf(mx, vals[k]); }
#pragma unroll
  for (int m = 1; m < 32; m <<= 1) mx = fmaxf(mx, __shfl_xor(mx, m, 32));
  if ((t & 31) == 0) red[t >> 5] = mx;
  __syncthreads();
  float bm = red[0];
#pragma unroll
  for (int i = 1; i < 8; ++i) bm = fmaxf(bm, red[i]);
  __syncthreads();
  float s = 0.f;
#pragma unroll
  for (int k = 0; k < 8; ++k) { vals[k] = __expf(vals[k] - bm); s += vals[k]; }
#pragma unroll
  for (int m = 1; m < 32; m <<= 1) s += __shfl_xor(s, m, 32);
  if ((t & 31) == 0) red[t >> 5] = s;
  __syncthreads();
  float bs = 0.f;
#pragma unroll
  for (int i = 0; i < 8; ++i) bs += red[i];
  float inv = 1.0f / bs;
#pragma unroll
  for (int k = 0; k < 8; ++k) wout[(size_t)b * L + t + k * 256] = vals[k] * inv;
}

// ---------- weighted sums (memory bound, coalesced over h/p) ----------
__global__ __launch_bounds__(256)
void k_rep1(const float* __restrict__ first, const float* __restrict__ wf,
            float* __restrict__ out) {
  int b = blockIdx.x >> 2;
  int h = ((blockIdx.x & 3) << 8) + threadIdx.x;
  const float* f = first + (size_t)b * L * HIDDEN + h;
  const float* w = wf + (size_t)b * L;
  float acc = 0.f;
  for (int i = 0; i < L; ++i) acc += w[i] * f[(size_t)i * HIDDEN];
  out[(size_t)b * HIDDEN + h] = acc;
}

__global__ __launch_bounds__(256)
void k_rep2(const float* __restrict__ second, const float* __restrict__ wsec,
            float* __restrict__ out) {
  int b = blockIdx.x;
  int p = threadIdx.x;
  if (p >= PROJ) return;
  const float* s = second + (size_t)b * L * PROJ + p;
  const float* w = wsec + (size_t)b * L;
  float acc = 0.f;
  for (int j = 0; j < L; ++j) acc += w[j] * s[(size_t)j * PROJ];
  out[(size_t)b * PROJ + p] = acc;
}

// ---------- launcher ----------
extern "C" void kernel_launch(void* const* d_in, const int* in_sizes, int n_in,
                              void* d_out, int out_size, void* d_ws, size_t ws_size,
                              hipStream_t stream) {
  (void)in_sizes; (void)n_in; (void)out_size; (void)ws_size;
  const float* first  = (const float*)d_in[0];   // [16,2048,1024]
  const float* second = (const float*)d_in[1];   // [16,2048,175]
  const float* param  = (const float*)d_in[2];   // [1024,175]
  float* out = (float*)d_out;

  // workspace layout (bytes)
  char* ws = (char*)d_ws;
  unsigned short* projb = (unsigned short*)(ws);                 // 16*2048*192*2 = 12,582,912
  unsigned short* secb  = (unsigned short*)(ws + 12582912);      // 12,582,912
  unsigned short* pT    = (unsigned short*)(ws + 25165824);      // 192*1024*2 = 393,216
  float* attn_i = (float*)(ws + 25559040);                       // 16*2048*4 = 131,072
  float* attn_j = (float*)(ws + 25690112);                       // 131,072 (contiguous w/ attn_i)

  // output layout (flat, return order)
  float* rep1 = out;             // [16,1024]
  float* wf   = out + 16384;     // [16,2048]
  float* rep2 = out + 49152;     // [16,175]
  float* wsec = out + 51952;     // [16,2048]

  k_paramT<<<(PPAD * HIDDEN) / 256, 256, 0, stream>>>(param, pT);
  k_sec<<<(NB * L * PPAD) / 256, 256, 0, stream>>>(second, secb);
  k_init_neginf<<<(2 * NB * L + 255) / 256, 256, 0, stream>>>(attn_i, 2 * NB * L);
  // proj: 16 * 128 i-tiles * 3 n-groups waves, 8 waves/block
  k_proj<<<(NB * (L / 16) * 3) / 8, 256, 0, stream>>>(first, pT, projb);
  // score: 16 * 128 ti * 32 tj-groups waves, 8 waves/block
  k_score<<<(NB * (L / 16) * (L / 64)) / 8, 256, 0, stream>>>(projb, secb, attn_i, attn_j);
  k_softmax<<<NB, 256, 0, stream>>>(attn_i, wf);
  k_softmax<<<NB, 256, 0, stream>>>(attn_j, wsec);
  k_rep1<<<NB * 4, 256, 0, stream>>>(first, wf, rep1);
  k_rep2<<<NB, 256, 0, stream>>>(second, wsec, rep2);
}
